// DynamicCylinderFeatureNet_26834955665463
// MI455X (gfx1250) — matrix operand order repaired and checked by reading, not compiled, not run on hardware
//
#include <hip/hip_runtime.h>
#include <hip/hip_bf16.h>

typedef __attribute__((ext_vector_type(16))) _Float16 v16h;
typedef __attribute__((ext_vector_type(8)))  float    v8f;

#define N_PTS 600000
#define N_VOX 150000
#define BN_EPS 1e-3f

// post-ReLU values are >= 0, so uint bit-pattern compare == float compare
__device__ __forceinline__ void atomic_max_relu(float* p, float v) {
    atomicMax(reinterpret_cast<unsigned int*>(p), __float_as_uint(v));
}

// ---------------- weight fp32 -> f16, pre-swizzled into WMMA B-fragment layout ----------
// Fragment (ct, kt): lane holds halves h=0..15 with K = kt*32 + (h<8 ? h : h+8) + (lane>=16 ? 8 : 0),
// N = ct*16 + (lane&15).  We store lane's 16 halves contiguously (32 B) so the GEMM
// kernels fetch a whole B fragment with two global_load_b128.
template <int KDIM, int ODIM>
__global__ __launch_bounds__(256)
void k_swz(const float* __restrict__ W, _Float16* __restrict__ dst)
{
    constexpr int NK = KDIM / 32;
    int i = blockIdx.x * blockDim.x + threadIdx.x;
    if (i >= KDIM * ODIM) return;
    int k = i / ODIM, n = i % ODIM;
    int ct = n >> 4, n0 = n & 15;
    int kt = k >> 5, kk = k & 31;
    int hi = (kk >> 3) & 1;
    int h  = (kk & 7) + ((kk >= 16) ? 8 : 0);
    int lane = n0 + hi * 16;
    dst[(((size_t)(ct * NK + kt)) * 32 + lane) * 16 + h] = (_Float16)W[i];
}

// ---------------- generic f32 -> f16 compaction (vf buffers between stages) ----------------
__global__ __launch_bounds__(256)
void k_f32_to_f16(const float* __restrict__ src, _Float16* __restrict__ dst, int n)
{
    int i = blockIdx.x * blockDim.x + threadIdx.x;
    if (i < n) dst[i] = (_Float16)src[i];
}

// ---------------- stage 0: build 9-dim feature, 9x64 GEMM + BN + ReLU + seg-max ----------------
__global__ __launch_bounds__(256)
void k_stage0(const float* __restrict__ feat, const int* __restrict__ inv,
              const int* __restrict__ coors, const float* __restrict__ W0,
              const float* __restrict__ g0, const float* __restrict__ b0,
              const float* __restrict__ m0, const float* __restrict__ v0,
              _Float16* __restrict__ p0, float* __restrict__ vf0)
{
    __shared__ float sW[9 * 64];
    __shared__ float sScale[64], sBias[64], sMean[64];
    for (int t = threadIdx.x; t < 9 * 64; t += blockDim.x) sW[t] = W0[t];
    for (int t = threadIdx.x; t < 64; t += blockDim.x) {
        sScale[t] = g0[t] * rsqrtf(v0[t] + BN_EPS);
        sBias[t]  = b0[t];
        sMean[t]  = m0[t];
    }
    __syncthreads();
    int i = blockIdx.x * blockDim.x + threadIdx.x;
    if (i >= N_PTS) return;

    float px = feat[i * 6 + 0], py = feat[i * 6 + 1], pz = feat[i * 6 + 2];
    int vi = inv[i];
    float c1 = (float)coors[vi * 4 + 1];
    float c2 = (float)coors[vi * 4 + 2];
    float c3 = (float)coors[vi * 4 + 3];
    const float vs0 = 0.1875f, vs1 = 0.017453292f, vs2 = 0.104166664f;
    float x[9];
    x[0] = px; x[1] = py; x[2] = pz;
    x[3] = px - ((c1 + 0.5f) * vs0 - 4.0f);
    x[4] = py - ((c2 + 0.5f) * vs1 - 3.14159265f);
    x[5] = pz - ((c3 + 0.5f) * vs2 + 0.0f);
    x[6] = feat[i * 6 + 3]; x[7] = feat[i * 6 + 4]; x[8] = feat[i * 6 + 5];

    #pragma unroll 4
    for (int o = 0; o < 64; o++) {
        float acc = 0.f;
        #pragma unroll
        for (int j = 0; j < 9; j++) acc += x[j] * sW[j * 64 + o];
        float h = (acc - sMean[o]) * sScale[o] + sBias[o];
        h = fmaxf(h, 0.f);
        p0[(long)i * 64 + o] = (_Float16)h;
        atomic_max_relu(&vf0[(long)vi * 64 + o], h);
    }
}

// ---------------- WMMA PFN stage: x = [p_in | vf_in[inv]]; p_out = relu(BN(x @ W)) ----------------
// One wave per 16-row block. A fragments per ISA 16-bit A layout; B pre-swizzled (32 B/lane/frag).
template <int KDIM, int ODIM, bool STORE_P>
__global__ __launch_bounds__(128)
void k_pfn_wmma(const _Float16* __restrict__ pin,  // N x (KDIM/2) f16
                const _Float16* __restrict__ vfh,  // V x (KDIM/2) f16 (compacted vf)
                const int* __restrict__ inv,
                const _Float16* __restrict__ Wswz, // swizzled B fragments
                const float* __restrict__ g, const float* __restrict__ b,
                const float* __restrict__ mm, const float* __restrict__ vv,
                _Float16* __restrict__ pout,       // N x ODIM f16 (if STORE_P)
                float* __restrict__ vfout)         // V x ODIM f32
{
    constexpr int KH = KDIM / 2;
    constexpr int NK = KDIM / 32;
    const int lane = threadIdx.x & 31;
    const int wave = threadIdx.x >> 5;
    const long rowbase = ((long)blockIdx.x * (blockDim.x >> 5) + wave) * 16;
    const long r = rowbase + (lane & 15);
    const int voxr = inv[r];
    const bool hi = lane >= 16;

    // Build all A fragments once (reused across every output column tile)
    v16h A[NK];
    #pragma unroll
    for (int kt = 0; kt < NK; kt++) {
        #pragma unroll
        for (int h = 0; h < 16; h++) {
            int k = kt * 32 + ((h < 8) ? (h + (hi ? 8 : 0)) : (h + 8 + (hi ? 8 : 0)));
            _Float16 val = (k < KH) ? pin[r * KH + k]
                                    : vfh[(long)voxr * KH + (k - KH)];
            A[kt][h] = val;
        }
    }

    const int n0 = lane & 15;
    #pragma unroll 1
    for (int ct = 0; ct < ODIM / 16; ct++) {
        v8f acc = {};
        #pragma unroll
        for (int kt = 0; kt < NK; kt++) {
            const v16h B = *reinterpret_cast<const v16h*>(
                Wswz + ((size_t)(ct * NK + kt) * 32 + lane) * 16);
            acc = __builtin_amdgcn_wmma_f32_16x16x32_f16(
                false, A[kt], false, B, (short)0, acc, false, false);
        }
        // epilogue: BN + ReLU, store point features, segment-max into voxel buffer
        const int n = ct * 16 + n0;
        const float sc = g[n] * rsqrtf(vv[n] + BN_EPS);
        const float bb = b[n], mu = mm[n];
        #pragma unroll
        for (int j = 0; j < 8; j++) {
            long r2 = rowbase + j + (hi ? 8 : 0);
            int voxj = __shfl(voxr, j + (hi ? 8 : 0), 32);  // inv[r2], no reload
            float h = (acc[j] - mu) * sc + bb;
            h = fmaxf(h, 0.f);
            if (STORE_P) pout[r2 * ODIM + n] = (_Float16)h;
            atomic_max_relu(&vfout[(long)voxj * ODIM + n], h);
        }
    }
}

// ---------------- head: out = relu(vf2 @ Wp + bp), one WMMA column tile ----------------
__global__ __launch_bounds__(32)
void k_head(const float* __restrict__ vf2, const _Float16* __restrict__ Wpswz,
            const float* __restrict__ bp, float* __restrict__ out)
{
    const int lane = threadIdx.x & 31;
    const long rowbase = (long)blockIdx.x * 16;
    const long r = rowbase + (lane & 15);
    const bool hi = lane >= 16;
    v8f acc = {};
    #pragma unroll
    for (int kt = 0; kt < 8; kt++) {
        v16h A;
        #pragma unroll
        for (int h = 0; h < 16; h++) {
            int ka = kt * 32 + ((h < 8) ? (h + (hi ? 8 : 0)) : (h + 8 + (hi ? 8 : 0)));
            A[h] = (_Float16)vf2[r * 256 + ka];
        }
        const v16h B = *reinterpret_cast<const v16h*>(
            Wpswz + ((size_t)kt * 32 + lane) * 16);
        acc = __builtin_amdgcn_wmma_f32_16x16x32_f16(
            false, A, false, B, (short)0, acc, false, false);
    }
    const int n = lane & 15;
    const float bb = bp[n];
    #pragma unroll
    for (int j = 0; j < 8; j++) {
        long r2 = rowbase + j + (hi ? 8 : 0);
        out[r2 * 16 + n] = fmaxf(acc[j] + bb, 0.f);
    }
}

// ---------------- copy voxel_coors (int32 bits) into tail of d_out ----------------
__global__ __launch_bounds__(256)
void k_copy_coors(const int* __restrict__ coors, int* __restrict__ dst, int n)
{
    int i = blockIdx.x * blockDim.x + threadIdx.x;
    if (i < n) dst[i] = coors[i];
}

extern "C" void kernel_launch(void* const* d_in, const int* in_sizes, int n_in,
                              void* d_out, int out_size, void* d_ws, size_t ws_size,
                              hipStream_t stream)
{
    const float* feat  = (const float*)d_in[0];
    const int*   inv   = (const int*)d_in[1];
    const int*   coors = (const int*)d_in[2];
    const float* W0 = (const float*)d_in[3];
    const float* g0 = (const float*)d_in[4];
    const float* b0 = (const float*)d_in[5];
    const float* m0 = (const float*)d_in[6];
    const float* v0 = (const float*)d_in[7];
    const float* W1 = (const float*)d_in[8];
    const float* g1 = (const float*)d_in[9];
    const float* b1 = (const float*)d_in[10];
    const float* m1 = (const float*)d_in[11];
    const float* v1 = (const float*)d_in[12];
    const float* W2 = (const float*)d_in[13];
    const float* g2 = (const float*)d_in[14];
    const float* b2 = (const float*)d_in[15];
    const float* m2 = (const float*)d_in[16];
    const float* v2 = (const float*)d_in[17];
    const float* Wp = (const float*)d_in[18];
    const float* bp = (const float*)d_in[19];

    char* ws = (char*)d_ws;
    size_t off = 0;
    float* vf0 = (float*)(ws + off); off += (size_t)N_VOX * 64  * 4;
    float* vf1 = (float*)(ws + off); off += (size_t)N_VOX * 128 * 4;
    float* vf2 = (float*)(ws + off); off += (size_t)N_VOX * 256 * 4;
    const size_t vf_bytes = off;                       // zero-init region (seg-max identity)
    _Float16* p0   = (_Float16*)(ws + off); off += (size_t)N_PTS * 64  * 2;
    _Float16* p1   = (_Float16*)(ws + off); off += (size_t)N_PTS * 128 * 2;
    _Float16* vf0h = (_Float16*)(ws + off); off += (size_t)N_VOX * 64  * 2;
    _Float16* vf1h = (_Float16*)(ws + off); off += (size_t)N_VOX * 128 * 2;
    _Float16* W1s  = (_Float16*)(ws + off); off += (size_t)128 * 128 * 2;
    _Float16* W2s  = (_Float16*)(ws + off); off += (size_t)256 * 256 * 2;
    _Float16* Wps  = (_Float16*)(ws + off); off += (size_t)256 * 16 * 2;

    hipMemsetAsync(ws, 0, vf_bytes, stream);

    // one-time weight convert + swizzle into B-fragment layout
    k_swz<128, 128><<<(128 * 128 + 255) / 256, 256, 0, stream>>>(W1, W1s);
    k_swz<256, 256><<<(256 * 256 + 255) / 256, 256, 0, stream>>>(W2, W2s);
    k_swz<256,  16><<<(256 * 16  + 255) / 256, 256, 0, stream>>>(Wp, Wps);

    k_stage0<<<(N_PTS + 255) / 256, 256, 0, stream>>>(
        feat, inv, coors, W0, g0, b0, m0, v0, p0, vf0);

    k_f32_to_f16<<<(N_VOX * 64 + 255) / 256, 256, 0, stream>>>(vf0, vf0h, N_VOX * 64);

    // 600000/16 = 37500 row-block waves; 4 waves / 128-thread block -> 9375 blocks
    k_pfn_wmma<128, 128, true><<<N_PTS / 64, 128, 0, stream>>>(
        p0, vf0h, inv, W1s, g1, b1, m1, v1, p1, vf1);

    k_f32_to_f16<<<(N_VOX * 128 + 255) / 256, 256, 0, stream>>>(vf1, vf1h, N_VOX * 128);

    k_pfn_wmma<256, 256, false><<<N_PTS / 64, 128, 0, stream>>>(
        p1, vf1h, inv, W2s, g2, b2, m2, v2, nullptr, vf2);

    // 150000/16 = 9375 voxel row blocks, one wave each
    k_head<<<N_VOX / 16, 32, 0, stream>>>(vf2, Wps, bp, (float*)d_out);

    int* coorout = (int*)d_out + (size_t)N_VOX * 16;
    k_copy_coors<<<(N_VOX * 4 + 255) / 256, 256, 0, stream>>>(coors, coorout, N_VOX * 4);
}
